// DGMNNet_87041807221283
// MI455X (gfx1250) — compile-verified, matrix operand order checked by reading, not run on hardware
//
#include <hip/hip_runtime.h>
#include <hip/hip_bf16.h>
#include <stdint.h>

// ---------------------------------------------------------------------------
// CDNA5 (gfx1250) wave32 WMMA types
// ---------------------------------------------------------------------------
typedef __attribute__((ext_vector_type(16))) __bf16 v16bf;
typedef __attribute__((ext_vector_type(8)))  float  v8f;

static __device__ __forceinline__ __bf16 f2bf(float f) {
  uint32_t u = __builtin_bit_cast(uint32_t, f);
  uint32_t r = u + 0x7FFFu + ((u >> 16) & 1u);   // round-to-nearest-even
  uint16_t h = (uint16_t)(r >> 16);
  return __builtin_bit_cast(__bf16, h);
}
static __device__ __forceinline__ float bf2f(__bf16 b) {
  uint16_t h = __builtin_bit_cast(uint16_t, b);
  uint32_t u = ((uint32_t)h) << 16;
  return __builtin_bit_cast(float, u);
}

// Problem constants
#define NB    2
#define CIN   512
#define HH    64
#define WWD   64
#define NPIX  8192        // 2*64*64
#define CCAT  3072        // 6*512
#define KCONV 4608        // 512*9

// ---------------------------------------------------------------------------
// Pack the 5 conv weights (27,512,3,3) OIHW fp32 -> bf16 [d][32][4608],
// k = (ky*3+kx)*512 + c, rows 27..31 zero-padded.
// ---------------------------------------------------------------------------
__global__ __launch_bounds__(256) void pack_weights(
    const float* __restrict__ w0, const float* __restrict__ w1,
    const float* __restrict__ w2, const float* __restrict__ w3,
    const float* __restrict__ w4, __bf16* __restrict__ wpk) {
  int idx = blockIdx.x * 256 + threadIdx.x;       // < 5*32*4608 = 737280
  int d  = idx / 147456;
  int r  = idx % 147456;
  int oc = r / 4608;
  int k  = r % 4608;
  int s  = k / 512;                                // ky*3+kx
  int c  = k % 512;
  const float* ws = (d == 0) ? w0 : (d == 1) ? w1 : (d == 2) ? w2
                                  : (d == 3) ? w3 : w4;
  float v = (oc < 27) ? ws[(oc * 512 + c) * 9 + s] : 0.f;
  wpk[idx] = f2bf(v);
}

// Generic fp32 -> bf16 copy (used for scale_w, already [oc][3072] in concat order)
__global__ __launch_bounds__(256) void convert_bf16(
    const float* __restrict__ src, __bf16* __restrict__ dst, int n) {
  int i = blockIdx.x * 256 + threadIdx.x;
  if (i < n) dst[i] = f2bf(src[i]);
}

// Zero-fill the 512-element bf16 "null pixel" used to absorb out-of-image taps
__global__ __launch_bounds__(256) void zero_null(__bf16* __restrict__ dst) {
  dst[threadIdx.x]       = f2bf(0.f);
  dst[threadIdx.x + 256] = f2bf(0.f);
}

// x NCHW fp32 -> concat buffer slot 0, NHWC bf16 with row stride 3072
__global__ __launch_bounds__(256) void pack_x(
    const float* __restrict__ x, __bf16* __restrict__ cat) {
  int idx = blockIdx.x * 256 + threadIdx.x;       // < 2*4096*512 = 4194304
  int pix = idx >> 9;
  int c   = idx & 511;
  int n   = pix >> 12;
  int hw  = pix & 4095;
  cat[(size_t)pix * CCAT + c] = f2bf(x[((size_t)(n * CIN + c)) * 4096 + hw]);
}

// ---------------------------------------------------------------------------
// Dilated 3x3 conv, 27 output channels, as implicit GEMM on matrix cores.
// Per wave: 16 pixels x 32 oc (two 16x16 C tiles), K-loop up to 9 slabs * 16
// csteps. Out-of-image rows (wave-uniform) skip the whole slab; out-of-image
// columns (per-lane) are redirected to a zeroed null pixel so the inner loop
// is branch-free: pure global_load_b128 + v_wmma.
// ---------------------------------------------------------------------------
__global__ __launch_bounds__(256) void conv27(
    const __bf16* __restrict__ cat, const __bf16* __restrict__ wpk,
    const __bf16* __restrict__ nullpix, float* __restrict__ fo, int dil) {
  int lane  = threadIdx.x & 31;
  int wave  = threadIdx.x >> 5;
  int lcol  = lane & 15;
  int lhalf = lane >> 4;
  int pixbase = blockIdx.x * 128 + wave * 16;     // 64 blocks * 8 waves * 16 pix
  int n  = pixbase >> 12;
  int h  = (pixbase >> 6) & 63;
  int w0 = pixbase & 63;
  int wcol = w0 + lcol;

  const __bf16* arow0 = wpk + (size_t)lcol * KCONV + lhalf * 16;        // oc 0..15
  const __bf16* arow1 = wpk + (size_t)(16 + lcol) * KCONV + lhalf * 16; // oc 16..31
  const __bf16* nrow  = nullpix + lhalf * 16;

  v8f acc0 = {};
  v8f acc1 = {};

  for (int s = 0; s < 9; ++s) {
    int ky = s / 3, kx = s % 3;
    int iy = h + (ky - 1) * dil;
    if ((unsigned)iy >= (unsigned)HH) continue;   // wave-uniform: slab is all-zero
    int ix = wcol + (kx - 1) * dil;
    bool vx = (unsigned)ix < (unsigned)WWD;       // per-lane column validity
    const __bf16* brow =
        vx ? cat + ((size_t)((n * HH + iy) * WWD + ix)) * CCAT + lhalf * 16
           : nrow;
    const __bf16* aslab0 = arow0 + s * 512;
    const __bf16* aslab1 = arow1 + s * 512;
#pragma unroll 4
    for (int cs = 0; cs < 512; cs += 32) {
      v16bf a0 = *(const v16bf*)(aslab0 + cs);
      v16bf a1 = *(const v16bf*)(aslab1 + cs);
      v16bf b  = *(const v16bf*)(brow + cs);
      acc0 = __builtin_amdgcn_wmma_f32_16x16x32_bf16(
          false, a0, false, b, (short)0, acc0, false, false);
      acc1 = __builtin_amdgcn_wmma_f32_16x16x32_bf16(
          false, a1, false, b, (short)0, acc1, false, false);
    }
  }

  // C tile: VGPR r, lane l -> oc = r + 8*(l/16), pixel = tilebase + l%16
  int pix = pixbase + lcol;
  int mrow = lhalf * 8;
#pragma unroll
  for (int r = 0; r < 8; ++r) {
    int oc0 = r + mrow;            // 0..15, always < 27
    fo[(size_t)pix * 27 + oc0] = acc0[r];
    int oc1 = 16 + oc0;            // 16..31
    if (oc1 < 27) fo[(size_t)pix * 27 + oc1] = acc1[r];
  }
}

// ---------------------------------------------------------------------------
// Deformable unfold + softmax-weighted sum. One workgroup per pixel.
// The 27 per-pixel logits/offsets are staged to LDS with CDNA5 async
// global->LDS loads (ASYNCcnt), then broadcast to all lanes.
// Channel-parallel bilinear gathers are coalesced (NHWC bf16 image).
// Writes concat slot `slot` (1..5).
// ---------------------------------------------------------------------------
__global__ __launch_bounds__(256) void deform_kernel(
    const __bf16* __restrict__ cat, const float* __restrict__ fo,
    __bf16* __restrict__ catout, int dil, int slot) {
  __shared__ float fos[27];
  int pix = blockIdx.x;
  int n = pix >> 12;
  int h = (pix >> 6) & 63;
  int w = pix & 63;

  // gfx1250 async copy global -> LDS, bypassing VGPRs (tracked by ASYNCcnt).
  if (threadIdx.x < 27) {
    const float* gsrc = fo + (size_t)pix * 27 + threadIdx.x;
    // as(3) pointer value == LDS byte offset (generic shared addr low 32 bits)
    uint32_t ldsoff = (uint32_t)(uintptr_t)
        ((__attribute__((address_space(3))) float*)&fos[threadIdx.x]);
    asm volatile("global_load_async_to_lds_b32 %0, %1, off"
                 :: "v"(ldsoff), "v"(gsrc) : "memory");
  }
  asm volatile("s_wait_asynccnt 0" ::: "memory");
  __syncthreads();

  // softmax over the 9 filter logits (broadcast math, every lane computes it)
  float m = fos[0];
#pragma unroll
  for (int k = 1; k < 9; ++k) m = fmaxf(m, fos[k]);
  float e[9], sum = 0.f;
#pragma unroll
  for (int k = 0; k < 9; ++k) { e[k] = __expf(fos[k] - m); sum += e[k]; }
  float inv = 1.f / sum;

  int   y0i[9], x0i[9];
  float w00[9], w01[9], w10[9], w11[9], filt[9];
#pragma unroll
  for (int k = 0; k < 9; ++k) {
    filt[k] = e[k] * inv;
    int ky = k / 3, kx = k % 3;
    float ys = (float)(h + (ky - 1) * dil) + fos[9 + 2 * k];   // dy: ch 9,11,...
    float xs = (float)(w + (kx - 1) * dil) + fos[10 + 2 * k];  // dx: ch 10,12,...
    float y0f = floorf(ys), x0f = floorf(xs);
    y0i[k] = (int)y0f; x0i[k] = (int)x0f;
    float wy1 = ys - y0f, wx1 = xs - x0f;
    float wy0 = 1.f - wy1, wx0 = 1.f - wx1;
    w00[k] = wy0 * wx0; w01[k] = wy0 * wx1;
    w10[k] = wy1 * wx0; w11[k] = wy1 * wx1;
  }

  const __bf16* img = cat + (size_t)n * 4096 * CCAT;
#pragma unroll
  for (int ci = 0; ci < 2; ++ci) {
    int c = threadIdx.x + ci * 256;
    float acc = 0.f;
#pragma unroll
    for (int k = 0; k < 9; ++k) {
      int y0 = y0i[k], x0 = x0i[k];   // block-uniform -> cheap scalar branches
      float v00 = 0.f, v01 = 0.f, v10 = 0.f, v11 = 0.f;
      bool xa = (x0 >= 0) & (x0 < WWD);
      bool xb = (x0 + 1 >= 0) & (x0 + 1 < WWD);
      if (y0 >= 0 && y0 < HH) {
        if (xa) v00 = bf2f(img[((size_t)(y0 * WWD + x0)) * CCAT + c]);
        if (xb) v01 = bf2f(img[((size_t)(y0 * WWD + x0 + 1)) * CCAT + c]);
      }
      if (y0 + 1 >= 0 && y0 + 1 < HH) {
        if (xa) v10 = bf2f(img[((size_t)((y0 + 1) * WWD + x0)) * CCAT + c]);
        if (xb) v11 = bf2f(img[((size_t)((y0 + 1) * WWD + x0 + 1)) * CCAT + c]);
      }
      acc += filt[k] * (v00 * w00[k] + v01 * w01[k] + v10 * w10[k] + v11 * w11[k]);
    }
    catout[(size_t)pix * CCAT + slot * CIN + c] = f2bf(acc);
  }
}

// ---------------------------------------------------------------------------
// 1x1 conv: y[oc][pix] = sum_k scale_w[oc][k] * cat[pix][k]
// M=512, N=8192, K=3072. 64x64 block tile, 8 waves x 2 C-tiles each.
// A fragment shared across the wave's two N tiles.
// ---------------------------------------------------------------------------
__global__ __launch_bounds__(256) void gemm1x1(
    const __bf16* __restrict__ cat, const __bf16* __restrict__ swpk,
    float* __restrict__ y) {
  int lane  = threadIdx.x & 31;
  int wave  = threadIdx.x >> 5;
  int lcol  = lane & 15;
  int lhalf = lane >> 4;
  int msub  = wave & 3;
  int npair = wave >> 2;

  int mbase  = blockIdx.y * 64 + msub * 16;
  int nbase0 = blockIdx.x * 64 + npair * 32;
  int nbase1 = nbase0 + 16;

  const __bf16* arow  = swpk + (size_t)(mbase + lcol) * CCAT + lhalf * 16;
  const __bf16* brow0 = cat + (size_t)(nbase0 + lcol) * CCAT + lhalf * 16;
  const __bf16* brow1 = cat + (size_t)(nbase1 + lcol) * CCAT + lhalf * 16;

  v8f acc0 = {};
  v8f acc1 = {};

#pragma unroll 4
  for (int k = 0; k < CCAT; k += 32) {
    if (k + 512 < CCAT) {                       // warm GL2 ahead of the K stream
      __builtin_prefetch(brow0 + k + 512, 0, 1);
      __builtin_prefetch(brow1 + k + 512, 0, 1);
    }
    v16bf a  = *(const v16bf*)(arow + k);
    v16bf b0 = *(const v16bf*)(brow0 + k);
    v16bf b1 = *(const v16bf*)(brow1 + k);
    acc0 = __builtin_amdgcn_wmma_f32_16x16x32_bf16(
        false, a, false, b0, (short)0, acc0, false, false);
    acc1 = __builtin_amdgcn_wmma_f32_16x16x32_bf16(
        false, a, false, b1, (short)0, acc1, false, false);
  }

#pragma unroll
  for (int r = 0; r < 8; ++r) {
    int oc = mbase + r + lhalf * 8;
    y[(size_t)oc * NPIX + nbase0 + lcol] = acc0[r];
    y[(size_t)oc * NPIX + nbase1 + lcol] = acc1[r];
  }
}

// ---------------------------------------------------------------------------
// Per-channel mean/var over N*H*W=8192, normalize + affine, write NCHW fp32.
// One block per channel; values kept in registers between passes.
// ---------------------------------------------------------------------------
__global__ __launch_bounds__(256) void bn_norm(
    const float* __restrict__ y, const float* __restrict__ gamma,
    const float* __restrict__ beta, float* __restrict__ out) {
  __shared__ float rs[256];
  __shared__ float rs2[256];
  int c = blockIdx.x;
  int tid = threadIdx.x;

  float vals[32];
  float s = 0.f, s2 = 0.f;
#pragma unroll
  for (int i = 0; i < 32; ++i) {
    float v = y[(size_t)c * NPIX + tid + i * 256];
    vals[i] = v;
    s += v;
    s2 += v * v;
  }
  rs[tid] = s; rs2[tid] = s2;
  __syncthreads();
  for (int off = 128; off > 0; off >>= 1) {
    if (tid < off) { rs[tid] += rs[tid + off]; rs2[tid] += rs2[tid + off]; }
    __syncthreads();
  }
  float mean = rs[0] * (1.f / 8192.f);
  float var  = rs2[0] * (1.f / 8192.f) - mean * mean;
  float scale = rsqrtf(var + 1e-5f) * gamma[c];
  float bias  = beta[c] - mean * scale;

#pragma unroll
  for (int i = 0; i < 32; ++i) {
    int pix = tid + i * 256;
    int n = pix >> 12;
    int hw = pix & 4095;
    out[((size_t)(n * CIN + c)) * 4096 + hw] = vals[i] * scale + bias;
  }
}

// ---------------------------------------------------------------------------
// Host launcher
// ---------------------------------------------------------------------------
extern "C" void kernel_launch(void* const* d_in, const int* in_sizes, int n_in,
                              void* d_out, int out_size, void* d_ws, size_t ws_size,
                              hipStream_t stream) {
  const float* x       = (const float*)d_in[0];
  const float* w1      = (const float*)d_in[1];
  const float* w2      = (const float*)d_in[2];
  const float* w3      = (const float*)d_in[3];
  const float* w4      = (const float*)d_in[4];
  const float* w5      = (const float*)d_in[5];
  const float* scale_w = (const float*)d_in[6];
  const float* gamma   = (const float*)d_in[7];
  const float* beta    = (const float*)d_in[8];

  // Workspace layout (all offsets 256B aligned):
  //   cat  : bf16 [8192][3072]          = 50,331,648 B
  //   wpk  : bf16 [5][32][4608]         =  1,474,560 B
  //   swpk : bf16 [512][3072]           =  3,145,728 B
  //   fo   : f32  [8192][27]            =    884,736 B (reused per dilation)
  //   yb   : f32  [512][8192]           = 16,777,216 B
  //   null : bf16 [512]                 =      1,024 B (zeroed)
  char* ws = (char*)d_ws;
  __bf16* cat  = (__bf16*)(ws + 0);
  __bf16* wpk  = (__bf16*)(ws + 50331648);
  __bf16* swpk = (__bf16*)(ws + 51806208);
  float*  fo   = (float*)(ws + 54951936);
  float*  yb   = (float*)(ws + 55836672);
  __bf16* nullpix = (__bf16*)(ws + 72613888);

  pack_weights<<<2880, 256, 0, stream>>>(w1, w2, w3, w4, w5, wpk);
  convert_bf16<<<6144, 256, 0, stream>>>(scale_w, swpk, 512 * 3072);
  zero_null<<<1, 256, 0, stream>>>(nullpix);
  pack_x<<<16384, 256, 0, stream>>>(x, cat);

  const int dils[5] = {1, 6, 12, 24, 36};
  for (int d = 0; d < 5; ++d) {
    conv27<<<64, 256, 0, stream>>>(cat, wpk + (size_t)d * 32 * KCONV, nullpix,
                                   fo, dils[d]);
    deform_kernel<<<8192, 256, 0, stream>>>(cat, fo, cat, dils[d], d + 1);
  }

  gemm1x1<<<dim3(128, 8), 256, 0, stream>>>(cat, swpk, yb);
  bn_norm<<<512, 256, 0, stream>>>(yb, gamma, beta, (float*)d_out);
}
